// AttentionLayer_50689204027553
// MI455X (gfx1250) — matrix operand order repaired and checked by reading, not compiled, not run on hardware
//
#include <hip/hip_runtime.h>
#include <hip/hip_bf16.h>
#include <math.h>

// ---------------------------------------------------------------------------
// MI455X attention layer, compile-only tuning.
// Problem: B=2, L=S=2048, D=1024, H=16, E=64.
// HBM-bound (attn tensor = 537MB fp32 is a required output); all GEMMs use
// v_wmma_f32_16x16x32_bf16 (wave32 CDNA5 WMMA). Softmax is register-resident
// (1 read + 1 write of the attn tensor — the mandatory minimum).
// ---------------------------------------------------------------------------

typedef __attribute__((ext_vector_type(16))) __bf16 v16bf;
typedef __attribute__((ext_vector_type(8)))  float  v8f;

union FragU { unsigned u[8]; v16bf v; };

__device__ __forceinline__ unsigned short f32_to_bf16(float f) {
    unsigned u = __float_as_uint(f);
    unsigned rnd = 0x7FFFu + ((u >> 16) & 1u);   // round-to-nearest-even
    return (unsigned short)((u + rnd) >> 16);
}

// ---------------------------------------------------------------------------
// fp32 -> bf16 cast (streaming)
// ---------------------------------------------------------------------------
__global__ __launch_bounds__(256)
void cast_f32_bf16(const float* __restrict__ in, unsigned short* __restrict__ out, int n) {
    int i = blockIdx.x * 256 + threadIdx.x;
    if (i < n) out[i] = f32_to_bf16(in[i]);
}

// ---------------------------------------------------------------------------
// 64x64-tile GEMM: C[M,N] = A[M,K](bf16) * B[K,N](bf16) + bias
// 128 threads = 4 waves; wave w computes rows [w*16, w*16+16) x all 64 cols
// (4 accumulators). K stepped by 32 (one WMMA depth).
// A staged row-major in LDS; B staged transposed ([n][k]) so every lane's
// fragment loads are contiguous 32-bit reads per the CDNA5 16-bit layouts.
// ---------------------------------------------------------------------------
template <bool OUTF32>
__global__ __launch_bounds__(128)
void gemm_bf16_64x64(const unsigned short* __restrict__ A,
                     const unsigned short* __restrict__ Bm,
                     const float* __restrict__ bias,
                     void* __restrict__ Cout,
                     int M, int N, int K)
{
    __shared__ unsigned       ldsA[64 * 16];   // 64 rows x 32 k   (bf16 pairs)
    __shared__ unsigned short ldsBT[64 * 32];  // [n][k]
    const int tid  = threadIdx.x;
    const int lane = tid & 31;
    const int wave = tid >> 5;
    const int m0 = blockIdx.y * 64;
    const int n0 = blockIdx.x * 64;

    v8f acc[4] = {};
    const unsigned* Aui = (const unsigned*)A;
    const int Kp2 = K >> 1;

    for (int k0 = 0; k0 < K; k0 += 32) {
        // stage A tile: 64x32 bf16 = 1024 dwords
        for (int t = tid; t < 1024; t += 128) {
            int r = t >> 4, c = t & 15;
            ldsA[t] = Aui[(size_t)(m0 + r) * Kp2 + (k0 >> 1) + c];
        }
        // stage B tile transposed: 2048 ushorts
        for (int t = tid; t < 2048; t += 128) {
            int k = t >> 6, n = t & 63;
            ldsBT[n * 32 + k] = Bm[(size_t)(k0 + k) * N + n0 + n];
        }
        // prefetch next k-tile while this one is consumed (global_prefetch_b8)
        if (k0 + 32 < K) {
            __builtin_prefetch(&Aui[(size_t)(m0 + (tid >> 1)) * Kp2 + ((k0 + 32) >> 1)], 0, 1);
            __builtin_prefetch(&Bm[(size_t)(k0 + 32 + (tid >> 2)) * N + n0], 0, 1);
        }
        __syncthreads();

        // A fragment (16x32 bf16): lane m = lane&15; K pattern per ISA table
        FragU af;
        {
            int m  = wave * 16 + (lane & 15);
            int kb = (lane < 16) ? 0 : 8;
            #pragma unroll
            for (int j = 0; j < 8; ++j) {
                int k = kb + ((j < 4) ? 2 * j : 16 + 2 * (j - 4));
                af.u[j] = ldsA[m * 16 + (k >> 1)];
            }
        }
        const unsigned* ldsBTu = (const unsigned*)ldsBT;
        #pragma unroll
        for (int t = 0; t < 4; ++t) {
            FragU bf;
            int n  = t * 16 + (lane & 15);
            int kh = (lane < 16) ? 0 : 8;
            #pragma unroll
            for (int j = 0; j < 8; ++j)
                bf.u[j] = ldsBTu[n * 16 + (kh >> 1) + j + ((j < 4) ? 0 : 4)];
            acc[t] = __builtin_amdgcn_wmma_f32_16x16x32_bf16(
                false, af.v, false, bf.v, (short)0, acc[t], false, false);
        }
        __syncthreads();
    }

    // epilogue: C/D layout -> VGPR r: M = r (+8 for lanes 16-31), N = lane&15
    #pragma unroll
    for (int t = 0; t < 4; ++t) {
        int n = n0 + t * 16 + (lane & 15);
        float bv = bias ? bias[n] : 0.0f;
        #pragma unroll
        for (int r = 0; r < 8; ++r) {
            int m = m0 + wave * 16 + ((lane < 16) ? 0 : 8) + r;
            float val = acc[t][r] + bv;
            if (OUTF32) ((float*)Cout)[(size_t)m * N + n] = val;
            else ((unsigned short*)Cout)[(size_t)m * N + n] = f32_to_bf16(val);
        }
    }
}

// ---------------------------------------------------------------------------
// scores[b,h,l,s] = (Q[b,l,h,:] . K[b,s,h,:]) / sqrt(64), masked, fp32 out.
// Per block: one 64x64 tile of one head. K-dim = E = 64 (two WMMA steps).
// ---------------------------------------------------------------------------
__global__ __launch_bounds__(128)
void attn_scores(const unsigned short* __restrict__ Qp,
                 const unsigned short* __restrict__ Kp,
                 const unsigned char* __restrict__ mask,
                 float* __restrict__ attn,
                 int H, int L, int S)
{
    __shared__ unsigned short ldsQ[64 * 64];  // [l][e]
    __shared__ unsigned short ldsK[64 * 64];  // [s][e]
    const int tid  = threadIdx.x;
    const int lane = tid & 31;
    const int wave = tid >> 5;
    const int bh = blockIdx.z, b = bh / H, h = bh % H;
    const int l0 = blockIdx.y * 64, s0 = blockIdx.x * 64;
    const int D = H * 64;

    const unsigned* Qu = (const unsigned*)Qp;
    const unsigned* Ku = (const unsigned*)Kp;
    unsigned* ldsQu = (unsigned*)ldsQ;
    unsigned* ldsKu = (unsigned*)ldsK;
    for (int t = tid; t < 2048; t += 128) {     // 4096 bf16 each, as dwords
        int r = t >> 5, c = t & 31;
        ldsQu[t] = Qu[(((size_t)(b * L + l0 + r) * D + h * 64) >> 1) + c];
        ldsKu[t] = Ku[(((size_t)(b * S + s0 + r) * D + h * 64) >> 1) + c];
    }
    __syncthreads();

    v8f acc[4] = {};
    #pragma unroll
    for (int ks = 0; ks < 64; ks += 32) {
        FragU qf;
        int m  = wave * 16 + (lane & 15);
        int kb = (lane < 16) ? 0 : 8;
        #pragma unroll
        for (int j = 0; j < 8; ++j) {
            int k = ks + kb + ((j < 4) ? 2 * j : 16 + 2 * (j - 4));
            qf.u[j] = ldsQu[m * 32 + (k >> 1)];
        }
        #pragma unroll
        for (int t = 0; t < 4; ++t) {
            FragU kf;
            int n  = t * 16 + (lane & 15);
            int kh = ks + ((lane < 16) ? 0 : 8);
            #pragma unroll
            for (int j = 0; j < 8; ++j)
                kf.u[j] = ldsKu[n * 32 + (kh >> 1) + j + ((j < 4) ? 0 : 4)];
            acc[t] = __builtin_amdgcn_wmma_f32_16x16x32_bf16(
                false, qf.v, false, kf.v, (short)0, acc[t], false, false);
        }
    }

    const float scale = 0.125f;  // 1/sqrt(64)
    #pragma unroll
    for (int t = 0; t < 4; ++t) {
        int s = s0 + t * 16 + (lane & 15);
        #pragma unroll
        for (int r = 0; r < 8; ++r) {
            int l = l0 + wave * 16 + ((lane < 16) ? 0 : 8) + r;
            float v = acc[t][r] * scale;
            if (mask[(size_t)l * S + s]) v = -__builtin_inff();
            attn[((size_t)bh * L + l) * S + s] = v;
        }
    }
}

// ---------------------------------------------------------------------------
// register-resident in-place row softmax (S == 2048, 256 threads, 8/thread)
// exactly 1 read + 1 write of the row; wave32 shuffle reduce + 8-wave combine
// ---------------------------------------------------------------------------
__global__ __launch_bounds__(256)
void softmax_rows(float* __restrict__ attn, int S)
{
    __shared__ float red[8];
    const int tid  = threadIdx.x;
    const int lane = tid & 31;
    const int wave = tid >> 5;
    float* p = attn + (size_t)blockIdx.x * S;

    float v[8];
    #pragma unroll
    for (int i = 0; i < 8; ++i) v[i] = p[tid + i * 256];

    // row max
    float mx = v[0];
    #pragma unroll
    for (int i = 1; i < 8; ++i) mx = fmaxf(mx, v[i]);
    #pragma unroll
    for (int o = 16; o > 0; o >>= 1) mx = fmaxf(mx, __shfl_xor(mx, o, 32));
    if (lane == 0) red[wave] = mx;
    __syncthreads();
    mx = red[0];
    #pragma unroll
    for (int w = 1; w < 8; ++w) mx = fmaxf(mx, red[w]);
    __syncthreads();

    // exp + row sum
    float sum = 0.0f;
    #pragma unroll
    for (int i = 0; i < 8; ++i) { v[i] = __expf(v[i] - mx); sum += v[i]; }
    #pragma unroll
    for (int o = 16; o > 0; o >>= 1) sum += __shfl_xor(sum, o, 32);
    if (lane == 0) red[wave] = sum;
    __syncthreads();
    sum = 0.0f;
    #pragma unroll
    for (int w = 0; w < 8; ++w) sum += red[w];

    float inv = 1.0f / sum;
    #pragma unroll
    for (int i = 0; i < 8; ++i) p[tid + i * 256] = v[i] * inv;
}

// ---------------------------------------------------------------------------
// ctx[b,l,h,:] = sum_s attn[b,h,l,s] * V[b,s,h,:]   (GEMM [L,S]x[S,64], bf16 out)
// ---------------------------------------------------------------------------
__global__ __launch_bounds__(128)
void attn_av(const float* __restrict__ attn,
             const unsigned short* __restrict__ Vp,
             unsigned short* __restrict__ ctx,
             int H, int L, int S)
{
    __shared__ unsigned short ldsA[64 * 32];   // [l][k] bf16 (converted attn)
    __shared__ unsigned short ldsVT[64 * 32];  // [e][k]
    const int tid  = threadIdx.x;
    const int lane = tid & 31;
    const int wave = tid >> 5;
    const int bh = blockIdx.y, b = bh / H, h = bh % H;
    const int l0 = blockIdx.x * 64;
    const int D = H * 64;

    v8f acc[4] = {};
    const float* Arow = attn + (size_t)bh * L * S;

    for (int k0 = 0; k0 < S; k0 += 32) {
        for (int t = tid; t < 2048; t += 128) {
            int r = t >> 5, k = t & 31;
            ldsA[t] = f32_to_bf16(Arow[(size_t)(l0 + r) * S + k0 + k]);
        }
        for (int t = tid; t < 2048; t += 128) {
            int k = t >> 6, e = t & 63;
            ldsVT[e * 32 + k] = Vp[(size_t)(b * S + k0 + k) * D + h * 64 + e];
        }
        if (k0 + 32 < S) {
            __builtin_prefetch(&Arow[(size_t)(l0 + (tid >> 1)) * S + k0 + 32], 0, 1);
            __builtin_prefetch(&Vp[(size_t)(b * S + k0 + 32 + (tid >> 2)) * D + h * 64], 0, 1);
        }
        __syncthreads();

        FragU af;
        unsigned* ldsAu = (unsigned*)ldsA;
        unsigned* ldsVu = (unsigned*)ldsVT;
        int m  = wave * 16 + (lane & 15);
        int kb = (lane < 16) ? 0 : 8;
        #pragma unroll
        for (int j = 0; j < 8; ++j) {
            int k = kb + ((j < 4) ? 2 * j : 16 + 2 * (j - 4));
            af.u[j] = ldsAu[m * 16 + (k >> 1)];
        }
        #pragma unroll
        for (int t = 0; t < 4; ++t) {
            FragU bf;
            int n  = t * 16 + (lane & 15);
            int kh = (lane < 16) ? 0 : 8;
            #pragma unroll
            for (int j = 0; j < 8; ++j)
                bf.u[j] = ldsVu[n * 16 + (kh >> 1) + j + ((j < 4) ? 0 : 4)];
            acc[t] = __builtin_amdgcn_wmma_f32_16x16x32_bf16(
                false, af.v, false, bf.v, (short)0, acc[t], false, false);
        }
        __syncthreads();
    }

    #pragma unroll
    for (int t = 0; t < 4; ++t) {
        int e = t * 16 + (lane & 15);
        #pragma unroll
        for (int r = 0; r < 8; ++r) {
            int l = l0 + wave * 16 + ((lane < 16) ? 0 : 8) + r;
            ctx[(size_t)(b * L + l) * D + h * 64 + e] = f32_to_bf16(acc[t][r]);
        }
    }
}

// ---------------------------------------------------------------------------
// launch
// ---------------------------------------------------------------------------
extern "C" void kernel_launch(void* const* d_in, const int* in_sizes, int n_in,
                              void* d_out, int out_size, void* d_ws, size_t ws_size,
                              hipStream_t stream)
{
    constexpr int B = 2, L = 2048, S = 2048, D = 1024, H = 16;
    constexpr int M = B * L;                       // 4096 rows for projections
    constexpr size_t NQ = (size_t)B * L * D;       // 4,194,304
    constexpr size_t NW = (size_t)D * D;           // 1,048,576

    const float* queries = (const float*)d_in[0];
    const float* keys    = (const float*)d_in[1];
    const float* values  = (const float*)d_in[2];
    const unsigned char* mask = (const unsigned char*)d_in[3];
    const float* Wq = (const float*)d_in[4];  const float* bq = (const float*)d_in[5];
    const float* Wk = (const float*)d_in[6];  const float* bk = (const float*)d_in[7];
    const float* Wv = (const float*)d_in[8];  const float* bv = (const float*)d_in[9];
    const float* Wo = (const float*)d_in[10]; const float* bo = (const float*)d_in[11];

    float* out_proj = (float*)d_out;                 // (B,L,D) fp32
    float* attn     = (float*)d_out + NQ;            // (B,H,L,S) fp32, 2nd output

    // workspace layout (bf16 buffers), 64MB total
    char* ws = (char*)d_ws;
    unsigned short* qc  = (unsigned short*)(ws + 0);              // inputs bf16
    unsigned short* kc  = (unsigned short*)(ws + 8u  * 1048576);
    unsigned short* vc  = (unsigned short*)(ws + 16u * 1048576);
    unsigned short* wqc = (unsigned short*)(ws + 24u * 1048576);
    unsigned short* wkc = (unsigned short*)(ws + 26u * 1048576);
    unsigned short* wvc = (unsigned short*)(ws + 28u * 1048576);
    unsigned short* woc = (unsigned short*)(ws + 30u * 1048576);
    unsigned short* Qp  = (unsigned short*)(ws + 32u * 1048576);  // projected bf16
    unsigned short* Kp  = (unsigned short*)(ws + 40u * 1048576);
    unsigned short* Vp  = (unsigned short*)(ws + 48u * 1048576);
    unsigned short* ctx = (unsigned short*)(ws + 56u * 1048576);

    // 1) cast inputs + weights to bf16
    cast_f32_bf16<<<(NQ + 255) / 256, 256, 0, stream>>>(queries, qc, (int)NQ);
    cast_f32_bf16<<<(NQ + 255) / 256, 256, 0, stream>>>(keys,    kc, (int)NQ);
    cast_f32_bf16<<<(NQ + 255) / 256, 256, 0, stream>>>(values,  vc, (int)NQ);
    cast_f32_bf16<<<(NW + 255) / 256, 256, 0, stream>>>(Wq, wqc, (int)NW);
    cast_f32_bf16<<<(NW + 255) / 256, 256, 0, stream>>>(Wk, wkc, (int)NW);
    cast_f32_bf16<<<(NW + 255) / 256, 256, 0, stream>>>(Wv, wvc, (int)NW);
    cast_f32_bf16<<<(NW + 255) / 256, 256, 0, stream>>>(Wo, woc, (int)NW);

    // 2) QKV projections (WMMA GEMM, bf16 out)
    dim3 gProj(D / 64, M / 64);
    gemm_bf16_64x64<false><<<gProj, 128, 0, stream>>>(qc, wqc, bq, Qp, M, D, D);
    gemm_bf16_64x64<false><<<gProj, 128, 0, stream>>>(kc, wkc, bk, Kp, M, D, D);
    gemm_bf16_64x64<false><<<gProj, 128, 0, stream>>>(vc, wvc, bv, Vp, M, D, D);

    // 3) scores = Q.K^T / sqrt(E), masked, fp32 into attn output region
    dim3 gScores(S / 64, L / 64, B * H);
    attn_scores<<<gScores, 128, 0, stream>>>(Qp, Kp, mask, attn, H, L, S);

    // 4) softmax rows in place (1R + 1W)
    softmax_rows<<<B * H * L, 256, 0, stream>>>(attn, S);

    // 5) ctx = attn @ V (WMMA, bf16 out)
    dim3 gAV(L / 64, B * H);
    attn_av<<<gAV, 128, 0, stream>>>(attn, Vp, ctx, H, L, S);

    // 6) out = ctx @ Wo + bo (WMMA, fp32 out)
    gemm_bf16_64x64<true><<<gProj, 128, 0, stream>>>(ctx, woc, bo, out_proj, M, D, D);
}
